// DeformableConv1D_24240795419540
// MI455X (gfx1250) — matrix operand order repaired and checked by reading, not compiled, not run on hardware
//
#include <hip/hip_runtime.h>

// Problem constants (from reference)
#define BB   16
#define LL   4096
#define CC   256
#define FF   512
#define KK   5
#define KC   (KK*CC)      // 1280
#define NPOS (BB*LL)      // 65536
#define NCHUNK (KC/32)    // 40 K-chunks of 32

#define BM   128          // positions per block
#define BN   128          // filters per block
#define ASTR 40           // LDS A row stride in bf16 elems (80 B, bank-spread, 16B aligned)
#define ABUF (BM*ASTR)    // elems per LDS A buffer (5120 -> 10240 B)

typedef __attribute__((ext_vector_type(16))) __bf16 v16bf;
typedef __attribute__((ext_vector_type(8)))  float  v8f;

union Frag {
    v16bf v;
    uint4 q[2];
};

__device__ __forceinline__ unsigned int f2bf(float f) {
    union { float f; unsigned int u; } v;
    v.f = f;
    unsigned int u = v.u;
    // round-to-nearest-even float32 -> bf16
    unsigned int r = u + 0x7FFFu + ((u >> 16) & 1u);
    return (r >> 16) & 0xFFFFu;
}

// ---------------------------------------------------------------------------
// Kernel 0a: x [B,L,C] fp32 -> xbf bf16 (same layout). Memory-bound prepass.
// ---------------------------------------------------------------------------
__global__ __launch_bounds__(256) void xconvert_kernel(
    const float* __restrict__ x, unsigned short* __restrict__ xbf) {
    size_t i = ((size_t)blockIdx.x * 256 + threadIdx.x) * 8;  // 16M elems total
    float4 f0 = *(const float4*)(x + i);
    float4 f1 = *(const float4*)(x + i + 4);
    uint4 o;
    o.x = f2bf(f0.x) | (f2bf(f0.y) << 16);
    o.y = f2bf(f0.z) | (f2bf(f0.w) << 16);
    o.z = f2bf(f1.x) | (f2bf(f1.y) << 16);
    o.w = f2bf(f1.z) | (f2bf(f1.w) << 16);
    *(uint4*)(xbf + i) = o;
}

// ---------------------------------------------------------------------------
// Kernel 0b: w_conv [K,C,F] fp32 -> Wt [F, K*C] bf16 (transposed for B frags)
// ---------------------------------------------------------------------------
__global__ __launch_bounds__(256) void wconv_transpose_kernel(
    const float* __restrict__ w, unsigned short* __restrict__ Wt) {
    int i = blockIdx.x * 256 + threadIdx.x;      // over F*KC = 655360
    if (i >= FF * KC) return;
    int f  = i / KC;
    int kc = i - f * KC;
    Wt[i] = (unsigned short)f2bf(w[(size_t)kc * FF + f]);
}

// ---------------------------------------------------------------------------
// Kernel 1: offsets = conv1d_same(x, w_off); idx = (int)clip(l + off, 0, L-1)
// ---------------------------------------------------------------------------
__global__ __launch_bounds__(256) void offsets_idx_kernel(
    const float* __restrict__ x, const float* __restrict__ w_off,
    int* __restrict__ idxbuf) {
    __shared__ float sw[KK * CC * KK];           // [t][c][k] flat, 6400 floats
    int tid = threadIdx.x;
    for (int i = tid; i < KK * CC * KK; i += 256) sw[i] = w_off[i];
    __syncthreads();

    int p = blockIdx.x * 256 + tid;              // < 65536 exactly
    int b = p >> 12;
    int l = p & (LL - 1);
    const float* xb = x + (size_t)b * LL * CC;

    float off[KK] = {0.f, 0.f, 0.f, 0.f, 0.f};
    for (int t = 0; t < KK; ++t) {
        int ll = l + t - (KK / 2);
        if (ll < 0 || ll >= LL) continue;        // 'SAME' zero padding
        const float* xr = xb + (size_t)ll * CC;
        const float* wr = sw + t * CC * KK;
        for (int c = 0; c < CC; c += 4) {
            float4 xv = *(const float4*)(xr + c);
            const float* wp = wr + c * KK;
#pragma unroll
            for (int k = 0; k < KK; ++k) {
                off[k] += xv.x * wp[k] + xv.y * wp[KK + k] +
                          xv.z * wp[2 * KK + k] + xv.w * wp[3 * KK + k];
            }
        }
    }
    float lf = (float)l;
#pragma unroll
    for (int k = 0; k < KK; ++k) {
        float pos = lf + off[k];                              // DIL == 1
        pos = fminf(fmaxf(pos, 0.0f), (float)(LL - 1));
        idxbuf[(size_t)p * KK + k] = (int)pos;                // truncating cast
    }
}

// ---------------------------------------------------------------------------
// Async global->LDS copy of 32 bytes (per lane); same INST_OFFSET applies to
// both global source and LDS destination.
// ---------------------------------------------------------------------------
__device__ __forceinline__ void async_copy32(unsigned int ldsaddr,
                                             const unsigned short* gsrc) {
    asm volatile(
        "global_load_async_to_lds_b128 %0, %1, off\n\t"
        "global_load_async_to_lds_b128 %0, %1, off offset:16"
        :: "v"(ldsaddr), "v"(gsrc) : "memory");
}

__device__ __forceinline__ void wait_async_le2() {
    asm volatile("s_wait_asynccnt 0x2" ::: "memory");
}

// Launder a pointer through a register so LLVM cannot fold/hoist the serial
// address chain into dozens of live "base+const" registers.
#define OPAQUE_PTR(p) asm volatile("" : "+v"(p))

// ---------------------------------------------------------------------------
// Kernel 2: deformable gather + GEMM via v_wmma_f32_16x16x32_bf16
//   out[p, f] = sum_kc A[p, kc] * Wt[f, kc] + bias[f]
// Block: 256 thr = 8 waves; tile BM=128 x BN=128.
// Waves: 4 along M (32 rows -> 2 A frags), 2 along N (64 cols -> 4 B frags).
// 4 LDS A buffers, async DMA prefetch depth 2, fully unrolled K loop; the
// staging address is a single serial running pointer (one live reg pair).
// Step s: compute buf[s&3]; DMA chunk s+2 -> buf[(s+2)&3];
//         s_wait_asynccnt<=2 forces chunk s+1 complete (in-order), barrier.
// ---------------------------------------------------------------------------
__global__ __launch_bounds__(256) void deform_gemm_kernel(
    const unsigned short* __restrict__ xbf, const float* __restrict__ bias,
    const unsigned short* __restrict__ Wt, const int* __restrict__ idxbuf,
    float* __restrict__ out) {
    __shared__ unsigned short ldsA[4][ABUF];     // 4 x 10 KB

    int tid = threadIdx.x;
    int p0  = blockIdx.x * BM;                   // first position of tile
    int b   = p0 >> 12;                          // 4096 % 128 == 0 -> same batch
    const unsigned short* xbf_b = xbf + (size_t)b * LL * CC;

    int wid    = tid >> 5;
    int lane   = tid & 31;
    int wave_m = wid & 3;                        // rows wave_m*32 .. +31
    int wave_n = wid >> 2;                       // cols wave_n*64 .. +63
    int mlane  = lane & 15;
    int half   = lane >> 4;
    int n0     = blockIdx.y * BN + wave_n * 64;

    // staging coords: thread covers (row = tid>>1, 16 bf16 cols at seg)
    int srow = tid >> 1;
    int sseg = (tid & 1) << 4;                   // 0 or 16

    // Per-tap gather base addresses for this thread's staging row (regs only)
    const unsigned short* abase[KK];
#pragma unroll
    for (int t = 0; t < KK; ++t) {
        int xrow = idxbuf[(size_t)(p0 + srow) * KK + t];
        abase[t] = xbf_b + (size_t)xrow * CC + sseg;
    }

    // LDS staging destinations, one per buffer
    unsigned int sdst[4];
#pragma unroll
    for (int q = 0; q < 4; ++q)
        sdst[q] = (unsigned int)(uintptr_t)&ldsA[q][srow * ASTR + sseg];

    // Per-wave A-fragment read base
    // lane<16 -> K 0..7 & 16..23 ; lane>=16 -> K 8..15 & 24..31
    const unsigned short* arow =
        &ldsA[0][(wave_m * 32 + mlane) * ASTR + half * 8];

    const unsigned short* wcol = Wt + (size_t)(n0 + mlane) * KC + half * 8;

    v8f acc[2][4] = {};

    // ---- prologue: DMA chunks 0 and 1 (both tap 0) via the running pointer
    const unsigned short* gsrc = abase[0];
    async_copy32(sdst[0], gsrc);                 // chunk 0
    gsrc += 32; OPAQUE_PTR(gsrc);
    async_copy32(sdst[1], gsrc);                 // chunk 1
    gsrc += 32; OPAQUE_PTR(gsrc);                // now at chunk 2
    wait_async_le2();                            // chunk 0 complete
    __syncthreads();

#pragma unroll
    for (int s = 0; s < NCHUNK; ++s) {
        // 1) DMA chunk s+2 into buf[(s+2)&3]; advance pointer to chunk s+3.
        //    All conditions/taps are compile-time constants (full unroll).
        if (s + 2 < NCHUNK) {
            async_copy32(sdst[(s + 2) & 3], gsrc);
            if (s + 3 < NCHUNK) {
                if (((s + 3) & 7) == 0) gsrc = abase[(s + 3) >> 3];
                else                    gsrc += 32;
                OPAQUE_PTR(gsrc);
            }
        }

        // 2) A fragments (2 x 16 rows) from buf[s&3]
        Frag a[2];
        const unsigned short* ap = arow + (s & 3) * ABUF;
#pragma unroll
        for (int i = 0; i < 2; ++i) {
            a[i].q[0] = *(const uint4*)(ap + i * 16 * ASTR);
            a[i].q[1] = *(const uint4*)(ap + i * 16 * ASTR + 16);
        }

        // 3) B fragments: all offsets fold to immediates off one base
        Frag bf[4];
        const unsigned short* wbase = wcol + s * 32;
#pragma unroll
        for (int j = 0; j < 4; ++j) {
            const uint4* wp = (const uint4*)(wbase + (size_t)j * 16 * KC);
            bf[j].q[0] = wp[0];
            bf[j].q[1] = wp[2];
        }

        // 4) 8 WMMAs: 2 M-frags x 4 N-subtiles
#pragma unroll
        for (int i = 0; i < 2; ++i)
#pragma unroll
            for (int j = 0; j < 4; ++j)
                acc[i][j] = __builtin_amdgcn_wmma_f32_16x16x32_bf16(
                    false, a[i].v, false, bf[j].v, (short)0, acc[i][j],
                    false, false);

        // 5) oldest in-flight DMA (chunk s+1) complete + block barrier
        wait_async_le2();
        __syncthreads();
    }

    // Epilogue: C/D layout -> VGPR g holds (M = g + half*8, N = mlane)
#pragma unroll
    for (int i = 0; i < 2; ++i) {
        float* outp = out +
            (size_t)(p0 + wave_m * 32 + i * 16 + half * 8) * FF + n0 + mlane;
#pragma unroll
        for (int j = 0; j < 4; ++j) {
            float bv = bias[n0 + j * 16 + mlane];
#pragma unroll
            for (int g = 0; g < 8; ++g) {
                outp[(size_t)g * FF + j * 16] = acc[i][j][g] + bv;
            }
        }
    }
}

// ---------------------------------------------------------------------------
extern "C" void kernel_launch(void* const* d_in, const int* in_sizes, int n_in,
                              void* d_out, int out_size, void* d_ws, size_t ws_size,
                              hipStream_t stream) {
    const float* x      = (const float*)d_in[0];   // [16,4096,256]
    const float* w_off  = (const float*)d_in[1];   // [5,256,5]
    const float* w_conv = (const float*)d_in[2];   // [5,256,512]
    const float* b_conv = (const float*)d_in[3];   // [512]
    float* out = (float*)d_out;                    // [16,4096,512]

    // Workspace layout
    unsigned short* Wt  = (unsigned short*)d_ws;                         // 1.25 MB
    int*            idxb = (int*)((char*)d_ws + (size_t)FF * KC * 2);    // 1.25 MB
    unsigned short* xbf = (unsigned short*)((char*)d_ws +
                          (size_t)FF * KC * 2 + (size_t)NPOS * KK * 4);  // 32 MB

    xconvert_kernel<<<(NPOS * CC) / (256 * 8), 256, 0, stream>>>(x, xbf);
    wconv_transpose_kernel<<<(FF * KC + 255) / 256, 256, 0, stream>>>(w_conv, Wt);
    offsets_idx_kernel<<<NPOS / 256, 256, 0, stream>>>(x, w_off, idxb);
    deform_gemm_kernel<<<dim3(NPOS / BM, FF / BN), 256, 0, stream>>>(
        xbf, b_conv, Wt, idxb, out);
}
